// CompressiveEncoder_6433861009887
// MI455X (gfx1250) — compile-verified, matrix operand order checked by reading, not run on hardware
//
#include <hip/hip_runtime.h>
#include <cstdint>
#include <cstddef>

// ---------------- problem constants ----------------
#define NL   4
#define Bn   8
#define Tt   512
#define Dd   512
#define Hh   8
#define DHd  64
#define DFF_ 2048
#define CM   128
#define MM   512
#define LKV  1152          // CM + MM + Tt
#define RAT  4
#define SCALE_ 0.125f      // 64^-0.5

typedef __bf16 bf16;
typedef bf16  v16bf __attribute__((ext_vector_type(16)));
typedef float v8f   __attribute__((ext_vector_type(8)));

union Frag32 { v16bf v; uint4 u[2]; };

// =====================================================================
// Batched WMMA GEMM:  C[z] = A[z] (MxK, lda) * Bt[z]^T (Bt is NxK, ldb)
// (+bias).  z = b*Hd + h.  Per-wave 32x32 tile = 4 accumulators;
// 4 waves/block cover a 64x64 macro-tile.  All fragment loads are
// contiguous 16B b128 loads; K stepped by 32.
// Requires: M,N multiples of 32, K multiple of 32, rows 16B-aligned.
// =====================================================================
__global__ __launch_bounds__(128)
void wmma_gemm_k(const bf16* __restrict__ A, const bf16* __restrict__ Bt,
                 float* __restrict__ C, const float* __restrict__ bias,
                 int M, int N, int K,
                 long long lda, long long ldb, long long ldc,
                 int Hd,
                 long long sAb, long long sAh,
                 long long sBb, long long sBh,
                 long long sCb, long long sCh)
{
    const int z = blockIdx.z;
    const int b = z / Hd;
    const int h = z % Hd;
    A  += (long long)b * sAb + (long long)h * sAh;
    Bt += (long long)b * sBb + (long long)h * sBh;
    C  += (long long)b * sCb + (long long)h * sCh;

    const int wave = threadIdx.x >> 5;
    const int lane = threadIdx.x & 31;
    const int m0 = (blockIdx.y * 2 + (wave >> 1)) * 32;
    const int n0 = (blockIdx.x * 2 + (wave & 1)) * 32;
    if (m0 >= M || n0 >= N) return;          // wave-uniform

    const int half = lane >> 4;              // K-octet select
    const int nr   = lane & 15;              // M idx in A frags, N idx in B/C frags

    const bf16* a0 = A  + (long long)(m0 + nr) * lda;
    const bf16* a1 = A  + (long long)(m0 + 16 + nr) * lda;
    const bf16* b0 = Bt + (long long)(n0 + nr) * ldb;
    const bf16* b1 = Bt + (long long)(n0 + 16 + nr) * ldb;

    v8f c00 = {}, c01 = {}, c10 = {}, c11 = {};
#pragma unroll 2
    for (int k0 = 0; k0 < K; k0 += 32) {
        const int kb = k0 + 8 * half;
        Frag32 fa0, fa1, fb0, fb1;
        fa0.u[0] = *(const uint4*)(a0 + kb);
        fa0.u[1] = *(const uint4*)(a0 + kb + 16);
        fa1.u[0] = *(const uint4*)(a1 + kb);
        fa1.u[1] = *(const uint4*)(a1 + kb + 16);
        fb0.u[0] = *(const uint4*)(b0 + kb);
        fb0.u[1] = *(const uint4*)(b0 + kb + 16);
        fb1.u[0] = *(const uint4*)(b1 + kb);
        fb1.u[1] = *(const uint4*)(b1 + kb + 16);
        c00 = __builtin_amdgcn_wmma_f32_16x16x32_bf16(false, fa0.v, false, fb0.v, (short)0, c00, false, false);
        c01 = __builtin_amdgcn_wmma_f32_16x16x32_bf16(false, fa0.v, false, fb1.v, (short)0, c01, false, false);
        c10 = __builtin_amdgcn_wmma_f32_16x16x32_bf16(false, fa1.v, false, fb0.v, (short)0, c10, false, false);
        c11 = __builtin_amdgcn_wmma_f32_16x16x32_bf16(false, fa1.v, false, fb1.v, (short)0, c11, false, false);
    }

    const float bb0 = bias ? bias[n0 + nr] : 0.0f;
    const float bb1 = bias ? bias[n0 + 16 + nr] : 0.0f;
#pragma unroll
    for (int r = 0; r < 8; ++r) {
        const long long r0 = (long long)(m0 + r + 8 * half) * ldc;
        const long long r1 = (long long)(m0 + 16 + r + 8 * half) * ldc;
        C[r0 + n0 + nr]      = c00[r] + bb0;
        C[r0 + n0 + 16 + nr] = c01[r] + bb1;
        C[r1 + n0 + nr]      = c10[r] + bb0;
        C[r1 + n0 + 16 + nr] = c11[r] + bb1;
    }
}

// ---------------- elementwise / reduction kernels ----------------

__global__ void zero_f32_k(float* p, long long n) {
    long long i = (long long)blockIdx.x * blockDim.x + threadIdx.x;
    if (i < n) p[i] = 0.0f;
}

__global__ void cvt_bf16_k(const float* __restrict__ s, bf16* __restrict__ d, long long n) {
    long long i = (long long)blockIdx.x * blockDim.x + threadIdx.x;
    if (i < n) d[i] = (bf16)s[i];
}

// src (Bz, R, C) f32 row-major -> dst (Bz, C, R) bf16 (transpose + convert)
__global__ void cvt_bf16_T_k(const float* __restrict__ s, bf16* __restrict__ d,
                             int R, int C, long long n) {
    long long i = (long long)blockIdx.x * blockDim.x + threadIdx.x;
    if (i >= n) return;
    int c = (int)(i % C);
    long long t = i / C;
    int r = (int)(t % R);
    long long b = t / R;
    d[(b * C + c) * (long long)R + r] = (bf16)s[i];
}

// src: (B, rows, cols) contiguous -> dst[b*dstBatchStride + (rowOff+r)*cols + c]
__global__ void cvt_concat_k(const float* __restrict__ s, bf16* __restrict__ d,
                             int rows, int cols, long long dstBatchStride,
                             long long rowOff, long long n) {
    long long i = (long long)blockIdx.x * blockDim.x + threadIdx.x;
    if (i >= n) return;
    int c = (int)(i % cols);
    long long t = i / cols;
    int r = (int)(t % rows);
    int b = (int)(t / rows);
    d[(long long)b * dstBatchStride + (rowOff + r) * cols + c] = (bf16)s[i];
}

// conv_w (L, o, c, r) -> WcT (L, o, r*D + c)  bf16   (already N x K layout)
__global__ void conv_repack_k(const float* __restrict__ cw, bf16* __restrict__ dst, long long n) {
    long long i = (long long)blockIdx.x * blockDim.x + threadIdx.x;
    if (i >= n) return;
    int r = (int)(i % RAT);
    long long t = i / RAT;
    int c = (int)(t % Dd);  t /= Dd;
    int o = (int)(t % Dd);
    int l = (int)(t / Dd);
    dst[(((long long)l * Dd + o) * RAT * Dd) + (r * Dd + c)] = (bf16)cw[i];
}

__global__ void embed_pe_k(const int* __restrict__ seq, const float* __restrict__ emb,
                           float* __restrict__ x, long long n) {
    long long i = (long long)blockIdx.x * blockDim.x + threadIdx.x;
    if (i >= n) return;
    int d = (int)(i % Dd);
    long long bt = i / Dd;
    int t = (int)(bt % Tt);
    int tok = seq[bt];
    int pair = d >> 1;
    float div = __expf(-(float)(2 * pair) * (9.210340371976184f / (float)Dd)); // ln(10000)
    float ang = (float)t * div;
    float pe = (d & 1) ? __cosf(ang) : __sinf(ang);
    x[i] = emb[(long long)tok * Dd + d] + pe;
}

__global__ __launch_bounds__(256)
void layernorm_k(const float* __restrict__ x, const float* __restrict__ g,
                 const float* __restrict__ bb, float* __restrict__ outF,
                 bf16* __restrict__ outB) {
    __shared__ float red[256];
    const int row = blockIdx.x;
    const int tid = threadIdx.x;
    const float* xr = x + (long long)row * Dd;

    float s = 0.0f;
    for (int i = tid; i < Dd; i += 256) s += xr[i];
    red[tid] = s; __syncthreads();
    for (int o = 128; o > 0; o >>= 1) { if (tid < o) red[tid] += red[tid + o]; __syncthreads(); }
    float mu = red[0] / (float)Dd; __syncthreads();

    float v = 0.0f;
    for (int i = tid; i < Dd; i += 256) { float dd = xr[i] - mu; v += dd * dd; }
    red[tid] = v; __syncthreads();
    for (int o = 128; o > 0; o >>= 1) { if (tid < o) red[tid] += red[tid + o]; __syncthreads(); }
    float rstd = rsqrtf(red[0] / (float)Dd + 1e-5f);

    for (int i = tid; i < Dd; i += 256) {
        float o = (xr[i] - mu) * rstd * g[i] + bb[i];
        if (outF) outF[(long long)row * Dd + i] = o;
        outB[(long long)row * Dd + i] = (bf16)o;
    }
}

// row softmax in-place with scale; optional attention mask (uchar, per (b,t))
// valid(j) = (j < off) || (mask[b][i] && mask[b][j-off])
__global__ __launch_bounds__(256)
void softmax_k(float* __restrict__ data, int Cc, float scale,
               const unsigned char* __restrict__ mask, int off) {
    __shared__ float red[256];
    const long long row = blockIdx.x;
    const int tid = threadIdx.x;
    float* dr = data + row * (long long)Cc;

    bool qv = true;
    const unsigned char* mrow = nullptr;
    if (mask) {
        int b = (int)(row / ((long long)Hh * Tt));
        int iq = (int)(row % Tt);
        mrow = mask + (long long)b * Tt;
        qv = mrow[iq] != 0;
    }
    const float NEG = -3.402823466e38f;

    float mx = NEG;
    for (int j = tid; j < Cc; j += 256) {
        float v = dr[j] * scale;
        if (mask) {
            bool ok = (j < off) || (qv && (mrow[j - off] != 0));
            if (!ok) v = NEG;
        }
        dr[j] = v;
        mx = fmaxf(mx, v);
    }
    red[tid] = mx; __syncthreads();
    for (int o = 128; o > 0; o >>= 1) { if (tid < o) red[tid] = fmaxf(red[tid], red[tid + o]); __syncthreads(); }
    float m = red[0]; __syncthreads();

    float sum = 0.0f;
    for (int j = tid; j < Cc; j += 256) {
        float e = __expf(dr[j] - m);
        dr[j] = e;
        sum += e;
    }
    red[tid] = sum; __syncthreads();
    for (int o = 128; o > 0; o >>= 1) { if (tid < o) red[tid] += red[tid + o]; __syncthreads(); }
    float inv = 1.0f / red[0];
    for (int j = tid; j < Cc; j += 256) dr[j] *= inv;
}

// acc[idx] += sum_z probs[z*TL + idx]   (idx over T*L, z over B*H)
__global__ void attn_accum_k(const float* __restrict__ probs, float* __restrict__ acc,
                             int Z, long long TL) {
    long long i = (long long)blockIdx.x * blockDim.x + threadIdx.x;
    if (i >= TL) return;
    float s = 0.0f;
    for (int z = 0; z < Z; ++z) s += probs[(long long)z * TL + i];
    acc[i] += s;
}

__global__ void add_inplace_k(float* __restrict__ x, const float* __restrict__ y, long long n) {
    long long i = (long long)blockIdx.x * blockDim.x + threadIdx.x;
    if (i < n) x[i] += y[i];
}

__global__ void gelu_bf16_k(const float* __restrict__ s, bf16* __restrict__ d, long long n) {
    long long i = (long long)blockIdx.x * blockDim.x + threadIdx.x;
    if (i >= n) return;
    float v = s[i];
    d[i] = (bf16)(0.5f * v * (1.0f + erff(v * 0.7071067811865476f)));
}

__global__ __launch_bounds__(256)
void mse_accum_k(const float* __restrict__ a, const float* __restrict__ b,
                 float* __restrict__ acc, long long n, float scale) {
    __shared__ float red[256];
    const int tid = threadIdx.x;
    float s = 0.0f;
    for (long long i = (long long)blockIdx.x * blockDim.x + tid; i < n;
         i += (long long)gridDim.x * blockDim.x) {
        float d = a[i] - b[i];
        s += d * d;
    }
    red[tid] = s; __syncthreads();
    for (int o = 128; o > 0; o >>= 1) { if (tid < o) red[tid] += red[tid + o]; __syncthreads(); }
    if (tid == 0) atomicAdd(acc, red[0] * scale);
}

__global__ void finalize_k(const float* __restrict__ auxSum, const float* __restrict__ attnSum,
                           float* __restrict__ outAux, float* __restrict__ outZero,
                           float* __restrict__ outAttn, long long n, float invC) {
    long long i = (long long)blockIdx.x * blockDim.x + threadIdx.x;
    if (i < n) outAttn[i] = attnSum[i] * invC;
    if (i == 0) { outAux[0] = auxSum[0] * (1.0f / NL); outZero[0] = 0.0f; }
}

// ---------------- host side ----------------

static inline void gemm(hipStream_t s, const bf16* A, const bf16* Bt, float* C,
                        const float* bias, int M, int N, int K,
                        long long lda, long long ldb, long long ldc,
                        int batch, int Hd,
                        long long sAb, long long sAh,
                        long long sBb, long long sBh,
                        long long sCb, long long sCh) {
    dim3 g((N + 63) / 64, (M + 63) / 64, batch);
    wmma_gemm_k<<<g, 128, 0, s>>>(A, Bt, C, bias, M, N, K, lda, ldb, ldc,
                                  Hd, sAb, sAh, sBb, sBh, sCb, sCh);
}

#define EWG(n) dim3((unsigned)(((n) + 255) / 256)), dim3(256)

extern "C" void kernel_launch(void* const* d_in, const int* in_sizes, int n_in,
                              void* d_out, int out_size, void* d_ws, size_t ws_size,
                              hipStream_t stream) {
    (void)in_sizes; (void)n_in; (void)out_size; (void)ws_size;

    const int*           seq   = (const int*)d_in[0];
    const unsigned char* mask  = (const unsigned char*)d_in[1];
    const float* mems  = (const float*)d_in[2];
    const float* cmems = (const float*)d_in[3];
    const float* emb   = (const float*)d_in[4];
    const float* ln1g  = (const float*)d_in[5];
    const float* ln1b  = (const float*)d_in[6];
    const float* Wq    = (const float*)d_in[7];
    const float* Wkv   = (const float*)d_in[8];
    const float* Wo    = (const float*)d_in[9];
    const float* bo    = (const float*)d_in[10];
    const float* convw = (const float*)d_in[11];
    const float* convb = (const float*)d_in[12];
    const float* ln2g  = (const float*)d_in[13];
    const float* ln2b  = (const float*)d_in[14];
    const float* W1    = (const float*)d_in[15];
    const float* b1    = (const float*)d_in[16];
    const float* W2    = (const float*)d_in[17];
    const float* b2    = (const float*)d_in[18];

    const long long BTD  = (long long)Bn * Tt * Dd;           // 2097152
    const long long BLD  = (long long)Bn * LKV * Dd;          // kv_in elems
    const long long BL2D = (long long)Bn * LKV * 2 * Dd;      // kv elems
    const long long DOTS = (long long)Bn * Hh * Tt * LKV;     // attn logits
    const long long TL   = (long long)Tt * LKV;

    float* outX     = (float*)d_out;
    float* outMems  = outX + BTD;
    float* outCmems = outMems + (long long)NL * Bn * MM * Dd;
    float* outAux   = outCmems + (long long)NL * Bn * CM * Dd;
    float* outZero  = outAux + 1;
    float* outAttn  = outZero + 1;

    // -------- workspace allocator --------
    char* ws = (char*)d_ws;
    size_t cur = 0;
    auto alloc = [&](size_t bytes) -> char* {
        char* p = ws + cur;
        cur += (bytes + 255) & ~(size_t)255;
        return p;
    };
    bf16* WqT   = (bf16*)alloc((size_t)NL * Dd * Dd * 2);          // (l, n, k)
    bf16* WkvT  = (bf16*)alloc((size_t)NL * 2 * Dd * Dd * 2);
    bf16* WoT   = (bf16*)alloc((size_t)NL * Dd * Dd * 2);
    bf16* WcT   = (bf16*)alloc((size_t)NL * Dd * RAT * Dd * 2);
    bf16* W1T   = (bf16*)alloc((size_t)NL * DFF_ * Dd * 2);
    bf16* W2T   = (bf16*)alloc((size_t)NL * Dd * DFF_ * 2);
    bf16* xnBf  = (bf16*)alloc((size_t)BTD * 2);
    bf16* xn2Bf = (bf16*)alloc((size_t)BTD * 2);
    float* qF   = (float*)alloc((size_t)BTD * 4);
    bf16* qBf   = (bf16*)alloc((size_t)BTD * 2);
    bf16* kvInBf= (bf16*)alloc((size_t)BLD * 2);
    float* kvF  = (float*)alloc((size_t)BL2D * 4);
    bf16* kvBf  = (bf16*)alloc((size_t)BL2D * 2);                  // (b, j, 2D)
    bf16* kvT   = (bf16*)alloc((size_t)BL2D * 2);                  // (b, 2D, j)
    float* dots = (float*)alloc((size_t)DOTS * 4);
    bf16* attnBf= (bf16*)alloc((size_t)DOTS * 2);
    float* ctxF = (float*)alloc((size_t)BTD * 4);
    bf16* ctxBf = (bf16*)alloc((size_t)BTD * 2);
    float* woF  = (float*)alloc((size_t)BTD * 4);
    bf16* compBf= (bf16*)alloc((size_t)Bn * CM * Dd * 2);
    float* ckvF = (float*)alloc((size_t)Bn * CM * 2 * Dd * 4);
    bf16* ckvBf = (bf16*)alloc((size_t)Bn * CM * 2 * Dd * 2);      // (b, j, 2D)
    bf16* ckvT  = (bf16*)alloc((size_t)Bn * 2 * Dd * CM * 2);      // (b, 2D, j)
    float* s1   = (float*)alloc((size_t)Bn * Hh * Tt * MM * 4);
    bf16* s1Bf  = (bf16*)alloc((size_t)Bn * Hh * Tt * MM * 2);
    float* s2   = (float*)alloc((size_t)Bn * Hh * Tt * CM * 4);
    bf16* s2Bf  = (bf16*)alloc((size_t)Bn * Hh * Tt * CM * 2);
    float* o1   = (float*)alloc((size_t)Bn * Hh * Tt * DHd * 4);
    float* o2   = (float*)alloc((size_t)Bn * Hh * Tt * DHd * 4);
    float* ff1F = (float*)alloc((size_t)Bn * Tt * DFF_ * 4);
    bf16* ff1Bf = (bf16*)alloc((size_t)Bn * Tt * DFF_ * 2);
    float* ff2F = (float*)alloc((size_t)BTD * 4);
    float* attnSum = (float*)alloc((size_t)TL * 4);
    float* auxSum  = (float*)alloc(256);

    // -------- prologue --------
    zero_f32_k<<<EWG(TL), 0, stream>>>(attnSum, TL);
    zero_f32_k<<<EWG(1), 0, stream>>>(auxSum, 1);

    // weights -> bf16, transposed to N x K
    cvt_bf16_T_k<<<EWG((long long)NL * Dd * Dd), 0, stream>>>(Wq, WqT, Dd, Dd, (long long)NL * Dd * Dd);
    cvt_bf16_T_k<<<EWG((long long)NL * Dd * 2 * Dd), 0, stream>>>(Wkv, WkvT, Dd, 2 * Dd, (long long)NL * Dd * 2 * Dd);
    cvt_bf16_T_k<<<EWG((long long)NL * Dd * Dd), 0, stream>>>(Wo, WoT, Dd, Dd, (long long)NL * Dd * Dd);
    cvt_bf16_T_k<<<EWG((long long)NL * Dd * DFF_), 0, stream>>>(W1, W1T, Dd, DFF_, (long long)NL * Dd * DFF_);
    cvt_bf16_T_k<<<EWG((long long)NL * DFF_ * Dd), 0, stream>>>(W2, W2T, DFF_, Dd, (long long)NL * DFF_ * Dd);
    conv_repack_k<<<EWG((long long)NL * Dd * Dd * RAT), 0, stream>>>(convw, WcT, (long long)NL * Dd * Dd * RAT);

    embed_pe_k<<<EWG(BTD), 0, stream>>>(seq, emb, outX, BTD);

    // -------- layers --------
    for (int l = 0; l < NL; ++l) {
        const bf16*  WqL  = WqT  + (long long)l * Dd * Dd;
        const bf16*  WkvL = WkvT + (long long)l * 2 * Dd * Dd;
        const bf16*  WoL  = WoT  + (long long)l * Dd * Dd;
        const bf16*  WcL  = WcT  + (long long)l * Dd * RAT * Dd;
        const bf16*  W1L  = W1T  + (long long)l * DFF_ * Dd;
        const bf16*  W2L  = W2T  + (long long)l * Dd * DFF_;
        const float* memL  = mems  + (long long)l * Bn * MM * Dd;
        const float* cmemL = cmems + (long long)l * Bn * CM * Dd;
        float* newMemL  = outMems  + (long long)l * Bn * MM * Dd;   // == xn
        float* newCmemL = outCmems + (long long)l * Bn * CM * Dd;   // == compressed

        // 1) xn = LN1(x); f32 straight into new_mems slot, bf16 for GEMMs
        layernorm_k<<<dim3(Bn * Tt), dim3(256), 0, stream>>>(
            outX, ln1g + l * Dd, ln1b + l * Dd, newMemL, xnBf);

        // 2) kv_in (bf16): [cmems | mems | xn] per batch
        cvt_concat_k<<<EWG((long long)Bn * CM * Dd), 0, stream>>>(
            cmemL, kvInBf, CM, Dd, (long long)LKV * Dd, 0, (long long)Bn * CM * Dd);
        cvt_concat_k<<<EWG((long long)Bn * MM * Dd), 0, stream>>>(
            memL, kvInBf, MM, Dd, (long long)LKV * Dd, CM, (long long)Bn * MM * Dd);
        cvt_concat_k<<<EWG(BTD), 0, stream>>>(
            newMemL, kvInBf, Tt, Dd, (long long)LKV * Dd, CM + MM, BTD);

        // 3) q = xn @ Wq
        gemm(stream, xnBf, WqL, qF, nullptr, Bn * Tt, Dd, Dd, Dd, Dd, Dd,
             1, 1, 0, 0, 0, 0, 0, 0);
        cvt_bf16_k<<<EWG(BTD), 0, stream>>>(qF, qBf, BTD);

        // 4) kv = kv_in @ Wkv ; keep row-major bf16 and depth-major bf16
        gemm(stream, kvInBf, WkvL, kvF, nullptr, Bn * LKV, 2 * Dd, Dd, Dd, Dd, 2 * Dd,
             1, 1, 0, 0, 0, 0, 0, 0);
        cvt_bf16_k<<<EWG(BL2D), 0, stream>>>(kvF, kvBf, BL2D);
        cvt_bf16_T_k<<<EWG(BL2D), 0, stream>>>(kvF, kvT, LKV, 2 * Dd, BL2D);

        // 5) dots = Q . K^T  per (b,h)   (Bt rows = kv positions)
        gemm(stream, qBf, kvBf, dots, nullptr, Tt, LKV, DHd,
             Dd, 2 * Dd, LKV,
             Bn * Hh, Hh,
             (long long)Tt * Dd, DHd,
             (long long)LKV * 2 * Dd, DHd,
             (long long)Hh * Tt * LKV, (long long)Tt * LKV);

        // 6) masked scaled softmax (in-place)
        softmax_k<<<dim3(Bn * Hh * Tt), dim3(256), 0, stream>>>(dots, LKV, SCALE_, mask, CM + MM);

        // 7) accumulate attn probs for attns_mean
        attn_accum_k<<<EWG(TL), 0, stream>>>(dots, attnSum, Bn * Hh, TL);

        // 8) ctx = attn @ V   (Bt = depth-major V rows from kvT)
        cvt_bf16_k<<<EWG(DOTS), 0, stream>>>(dots, attnBf, DOTS);
        gemm(stream, attnBf, kvT + (long long)Dd * LKV, ctxF, nullptr, Tt, DHd, LKV,
             LKV, LKV, Dd,
             Bn * Hh, Hh,
             (long long)Hh * Tt * LKV, (long long)Tt * LKV,
             (long long)(2 * Dd) * LKV, (long long)DHd * LKV,
             (long long)Tt * Dd, DHd);
        cvt_bf16_k<<<EWG(BTD), 0, stream>>>(ctxF, ctxBf, BTD);

        // 9) x = ctx @ Wo + bo + x
        gemm(stream, ctxBf, WoL, woF, bo + l * Dd, Bn * Tt, Dd, Dd, Dd, Dd, Dd,
             1, 1, 0, 0, 0, 0, 0, 0);
        add_inplace_k<<<EWG(BTD), 0, stream>>>(outX, woF, BTD);

        // 10) compression conv == GEMM: (B x 128 x (4D)) @ WcT^T + conv_b
        gemm(stream, kvInBf + (long long)CM * Dd, WcL, newCmemL, convb + l * Dd,
             CM, Dd, RAT * Dd,
             RAT * Dd, RAT * Dd, Dd,
             Bn, 1,
             (long long)LKV * Dd, 0, 0, 0, (long long)CM * Dd, 0);
        cvt_bf16_k<<<EWG((long long)Bn * CM * Dd), 0, stream>>>(newCmemL, compBf, (long long)Bn * CM * Dd);

        // 11) ckv = compressed @ Wkv ; row-major + depth-major bf16
        gemm(stream, compBf, WkvL, ckvF, nullptr, Bn * CM, 2 * Dd, Dd, Dd, Dd, 2 * Dd,
             1, 1, 0, 0, 0, 0, 0, 0);
        cvt_bf16_k<<<EWG((long long)Bn * CM * 2 * Dd), 0, stream>>>(ckvF, ckvBf, (long long)Bn * CM * 2 * Dd);
        cvt_bf16_T_k<<<EWG((long long)Bn * CM * 2 * Dd), 0, stream>>>(ckvF, ckvT, CM, 2 * Dd, (long long)Bn * CM * 2 * Dd);

        // 12) aux attention #1: full_attn(q, k_mem, v_mem)
        gemm(stream, qBf, kvBf + (long long)CM * 2 * Dd, s1, nullptr, Tt, MM, DHd,
             Dd, 2 * Dd, MM,
             Bn * Hh, Hh,
             (long long)Tt * Dd, DHd,
             (long long)LKV * 2 * Dd, DHd,
             (long long)Hh * Tt * MM, (long long)Tt * MM);
        softmax_k<<<dim3(Bn * Hh * Tt), dim3(256), 0, stream>>>(s1, MM, SCALE_, nullptr, 0);
        cvt_bf16_k<<<EWG((long long)Bn * Hh * Tt * MM), 0, stream>>>(s1, s1Bf, (long long)Bn * Hh * Tt * MM);
        gemm(stream, s1Bf, kvT + (long long)Dd * LKV + CM, o1, nullptr, Tt, DHd, MM,
             MM, LKV, DHd,
             Bn * Hh, Hh,
             (long long)Hh * Tt * MM, (long long)Tt * MM,
             (long long)(2 * Dd) * LKV, (long long)DHd * LKV,
             (long long)Hh * Tt * DHd, (long long)Tt * DHd);

        // 13) aux attention #2: full_attn(q, ck, cv)
        gemm(stream, qBf, ckvBf, s2, nullptr, Tt, CM, DHd,
             Dd, 2 * Dd, CM,
             Bn * Hh, Hh,
             (long long)Tt * Dd, DHd,
             (long long)CM * 2 * Dd, DHd,
             (long long)Hh * Tt * CM, (long long)Tt * CM);
        softmax_k<<<dim3(Bn * Hh * Tt), dim3(256), 0, stream>>>(s2, CM, SCALE_, nullptr, 0);
        cvt_bf16_k<<<EWG((long long)Bn * Hh * Tt * CM), 0, stream>>>(s2, s2Bf, (long long)Bn * Hh * Tt * CM);
        gemm(stream, s2Bf, ckvT + (long long)Dd * CM, o2, nullptr, Tt, DHd, CM,
             CM, CM, DHd,
             Bn * Hh, Hh,
             (long long)Hh * Tt * CM, (long long)Tt * CM,
             (long long)(2 * Dd) * CM, (long long)DHd * CM,
             (long long)Hh * Tt * DHd, (long long)Tt * DHd);

        // 14) aux += mean((o1-o2)^2)
        {
            long long nA = (long long)Bn * Hh * Tt * DHd;
            mse_accum_k<<<dim3(512), dim3(256), 0, stream>>>(o1, o2, auxSum, nA, 1.0f / (float)nA);
        }

        // 15) FFN: x += gelu(LN2(x) @ W1 + b1) @ W2 + b2
        layernorm_k<<<dim3(Bn * Tt), dim3(256), 0, stream>>>(
            outX, ln2g + l * Dd, ln2b + l * Dd, nullptr, xn2Bf);
        gemm(stream, xn2Bf, W1L, ff1F, b1 + l * DFF_, Bn * Tt, DFF_, Dd, Dd, Dd, DFF_,
             1, 1, 0, 0, 0, 0, 0, 0);
        gelu_bf16_k<<<EWG((long long)Bn * Tt * DFF_), 0, stream>>>(ff1F, ff1Bf, (long long)Bn * Tt * DFF_);
        gemm(stream, ff1Bf, W2L, ff2F, b2 + l * Dd, Bn * Tt, Dd, DFF_, DFF_, DFF_, Dd,
             1, 1, 0, 0, 0, 0, 0, 0);
        add_inplace_k<<<EWG(BTD), 0, stream>>>(outX, ff2F, BTD);
    }

    // -------- epilogue: aux, zero, attns_mean --------
    finalize_k<<<EWG(TL), 0, stream>>>(auxSum, attnSum, outAux, outZero, outAttn,
                                       TL, 1.0f / (float)(NL * Bn * Hh));
}
